// Attention_49718541418642
// MI455X (gfx1250) — compile-verified, matrix operand order checked by reading, not run on hardware
//
#include <hip/hip_runtime.h>
#include <hip/hip_bf16.h>
#include <math.h>

typedef __bf16 bf16;
typedef __attribute__((ext_vector_type(16))) __bf16 v16bf;
typedef __attribute__((ext_vector_type(8)))  __bf16 v8bf;
typedef __attribute__((ext_vector_type(8)))  float   v8f;

#define Bb   4
#define Nn   2048
#define DIMd 1536
#define Hh   16
#define DHd  96
#define NHd  (Hh * DHd)     // 1536
#define QKVC (3 * NHd)      // 4608
#define Mrows (Bb * Nn)     // 8192

__device__ __constant__ const float kScale = 0.1020620726159658f; // 96^-0.5

// ---- workspace layout (bytes) ----
#define OFF_XB    ((size_t)0)                                   // x bf16 [B*N][DIM]
#define OFF_WQKVT (OFF_XB    + (size_t)Mrows * DIMd * 2)        // Wqkv^T bf16 [4608][1536]
#define OFF_WOUTT (OFF_WQKVT + (size_t)QKVC * DIMd * 2)         // Wout^T bf16 [1536][1536]
#define OFF_QKV   (OFF_WOUTT + (size_t)DIMd * DIMd * 2)         // qkv bf16 [B][3][H][N][DH]
#define OFF_AO    (OFF_QKV   + (size_t)Bb * 3 * Hh * Nn * DHd * 2) // attn out bf16 [B*N][H*DH]

// ------------------------------------------------------------------
// helpers
// ------------------------------------------------------------------
__device__ inline v8f v8f_zero() {
  v8f v;
#pragma unroll
  for (int i = 0; i < 8; ++i) v[i] = 0.0f;
  return v;
}

// Load one 16-bit WMMA operand fragment (v16bf) for this lane.
// p points at the start of this lane's row/column (32+ contiguous bf16).
// kb = (lane>=16) ? 8 : 0.  Elements are halves [kb..kb+7] and [kb+16..kb+23].
__device__ inline v16bf load_frag(const bf16* p, int kb) {
  union { v16bf v; v8bf h[2]; } u;
  u.h[0] = *(const v8bf*)(p + kb);
  u.h[1] = *(const v8bf*)(p + kb + 16);
  return u.v;
}

__device__ inline v8f wmma_bf16(v16bf a, v16bf b, v8f c) {
  return __builtin_amdgcn_wmma_f32_16x16x32_bf16(
      /*neg_a=*/false, a, /*neg_b=*/false, b,
      /*c_mod=*/(short)0, c, /*reuse_a=*/false, /*reuse_b=*/false);
}

// ------------------------------------------------------------------
// conversion kernels
// ------------------------------------------------------------------
__global__ void cvt_f32_to_bf16(const float* __restrict__ src,
                                bf16* __restrict__ dst, int count) {
  int i = blockIdx.x * blockDim.x + threadIdx.x;
  if (i < count) dst[i] = (bf16)src[i];
}

// src [rows][cols] f32 -> dst [cols][rows] bf16
__global__ void transpose_cvt(const float* __restrict__ src,
                              bf16* __restrict__ dst, int rows, int cols) {
  int i = blockIdx.x * blockDim.x + threadIdx.x;
  if (i >= rows * cols) return;
  int r = i / cols, c = i % cols;
  dst[(size_t)c * rows + r] = (bf16)src[(size_t)r * cols + c];
}

// ------------------------------------------------------------------
// tiled bf16 WMMA GEMM:  C[M][Ncols] = A[M][K] * Bt[Ncols][K]^T
// WG = 256 threads = 8 waves; WG tile 128(M) x 128(N);
// waves in 2(M) x 4(N) grid, each wave computes 64x32 (8 WMMAs / k-step).
// mode 0: write bf16 QKV into [B][3][H][N][DH] layout
// mode 1: write f32 (acc + bias[col]) into fout[M][Ncols]
// ------------------------------------------------------------------
__global__ __launch_bounds__(256) void gemm_bf16_kernel(
    const bf16* __restrict__ A, const bf16* __restrict__ Bt,
    int M, int K, int Ncols, int mode,
    bf16* __restrict__ qkv_out, float* __restrict__ fout,
    const float* __restrict__ bias) {
  __shared__ bf16 As[128][32];
  __shared__ bf16 Bs[128][32];

  const int t = threadIdx.x;
  const int lane = t & 31;
  const int wave = t >> 5;
  const int tiles_n = Ncols / 128;
  const int bm = blockIdx.x / tiles_n;
  const int bn = blockIdx.x % tiles_n;
  const int wm = wave & 1;   // 2 waves along M (64 rows each)
  const int wn = wave >> 1;  // 4 waves along N (32 cols each)
  const int kb = (lane >= 16) ? 8 : 0;

  v8f acc[4][2];
#pragma unroll
  for (int mt = 0; mt < 4; ++mt)
#pragma unroll
    for (int nt = 0; nt < 2; ++nt) acc[mt][nt] = v8f_zero();

  // per-thread staging coordinates: 128x32 tile = 512 16B chunks, 2/thread
  const int r0 = (t + 0)   >> 2, c0 = (t + 0)   & 3;
  const int r1 = (t + 256) >> 2, c1 = (t + 256) & 3;

  for (int k0 = 0; k0 < K; k0 += 32) {
    const bf16* ga = A  + (size_t)(bm * 128) * K + k0;
    const bf16* gb = Bt + (size_t)(bn * 128) * K + k0;
    // issue ALL four staging loads before any LDS store (distinct temps so
    // the compiler keeps them in flight simultaneously)
    v8bf ta0 = *(const v8bf*)(ga + (size_t)r0 * K + c0 * 8);
    v8bf ta1 = *(const v8bf*)(ga + (size_t)r1 * K + c1 * 8);
    v8bf tb0 = *(const v8bf*)(gb + (size_t)r0 * K + c0 * 8);
    v8bf tb1 = *(const v8bf*)(gb + (size_t)r1 * K + c1 * 8);
    // pull next k-slice toward cache while we compute (global_prefetch_b8)
    if (k0 + 32 < K) {
      __builtin_prefetch(ga + (size_t)r0 * K + 32, 0, 1);
      __builtin_prefetch(gb + (size_t)r0 * K + 32, 0, 1);
    }
    *(v8bf*)&As[r0][c0 * 8] = ta0;
    *(v8bf*)&As[r1][c1 * 8] = ta1;
    *(v8bf*)&Bs[r0][c0 * 8] = tb0;
    *(v8bf*)&Bs[r1][c1 * 8] = tb1;
    __syncthreads();

    v16bf af[4], bfr[2];
#pragma unroll
    for (int mt = 0; mt < 4; ++mt)
      af[mt] = load_frag(&As[wm * 64 + mt * 16 + (lane & 15)][0], kb);
#pragma unroll
    for (int nt = 0; nt < 2; ++nt)
      bfr[nt] = load_frag(&Bs[wn * 32 + nt * 16 + (lane & 15)][0], kb);

#pragma unroll
    for (int mt = 0; mt < 4; ++mt)
#pragma unroll
      for (int nt = 0; nt < 2; ++nt)
        acc[mt][nt] = wmma_bf16(af[mt], bfr[nt], acc[mt][nt]);

    __syncthreads();
  }

  // epilogue: C/D layout -> lane holds col (lane&15), vgpr r holds row r+8*(lane>=16)
#pragma unroll
  for (int mt = 0; mt < 4; ++mt)
#pragma unroll
    for (int nt = 0; nt < 2; ++nt)
#pragma unroll
      for (int r = 0; r < 8; ++r) {
        int m = bm * 128 + wm * 64 + mt * 16 + r + ((lane >> 4) << 3);
        int col = bn * 128 + wn * 32 + nt * 16 + (lane & 15);
        float v = acc[mt][nt][r];
        if (mode == 0) {
          int b = m >> 11, n = m & 2047;              // N = 2048
          int which = col / NHd, rem = col % NHd;     // q/k/v select
          int h = rem / DHd, dh = rem % DHd;
          qkv_out[((((size_t)b * 3 + which) * Hh + h) * Nn + n) * DHd + dh] =
              (bf16)v;
        } else {
          fout[(size_t)m * Ncols + col] = v + bias[col];
        }
      }
}

// ------------------------------------------------------------------
// RoPE on q,k in place: 3 chunks of 32 along DH, rotate-half per chunk.
// thread handles one (b,h,n,chunk,j) pair (j, j+16) for both q and k.
// ------------------------------------------------------------------
__global__ void rope_kernel(bf16* __restrict__ qkv,
                            const float* __restrict__ f1,
                            const float* __restrict__ f2,
                            const float* __restrict__ f3) {
  int idx = blockIdx.x * blockDim.x + threadIdx.x;
  const int total = Bb * Hh * Nn * 3 * 16;
  if (idx >= total) return;
  int j = idx % 16; idx /= 16;
  int c = idx % 3;  idx /= 3;
  int n = idx % Nn; idx /= Nn;
  int h = idx % Hh; idx /= Hh;
  int b = idx;
  const float* f = ((c == 0) ? f1 : (c == 1) ? f2 : f3) + (size_t)n * 32;
  float c0 = cosf(f[j]),      s0 = sinf(f[j]);
  float c1 = cosf(f[j + 16]), s1 = sinf(f[j + 16]);
#pragma unroll
  for (int which = 0; which < 2; ++which) {  // 0=q, 1=k
    bf16* p = qkv + ((((size_t)b * 3 + which) * Hh + h) * Nn + n) * DHd + c * 32;
    float x1 = (float)p[j], x2 = (float)p[j + 16];
    p[j]      = (bf16)(x1 * c0 - x2 * s0);
    p[j + 16] = (bf16)(x2 * c1 + x1 * s1);
  }
}

// ------------------------------------------------------------------
// flash attention: 128 threads = 4 waves, each wave owns 16 queries.
// grid = B*H*(N/64).  K tile 32x96 + V^T tile 96x32 staged in LDS.
// ------------------------------------------------------------------
__global__ __launch_bounds__(128) void attn_kernel(
    const bf16* __restrict__ qkv, bf16* __restrict__ ao) {
  __shared__ bf16 Ks[32][96];       // [key_local][dh]
  __shared__ bf16 Vt[96][32];       // [dh][key_local]
  __shared__ bf16 Ps[4][16][32];    // per-wave prob tile [m][key_local]

  const int lane = threadIdx.x & 31;
  const int wave = threadIdx.x >> 5;
  const int qblk = blockIdx.x & 31;        // N/64 = 32 blocks per (b,h)
  const int bh   = blockIdx.x >> 5;
  const int b = bh >> 4, h = bh & 15;
  const int q0 = qblk * 64;
  const int kb = (lane >= 16) ? 8 : 0;

  const bf16* qgl = qkv + ((((size_t)b * 3 + 0) * Hh + h) * Nn) * DHd;
  const bf16* kgl = qkv + ((((size_t)b * 3 + 1) * Hh + h) * Nn) * DHd;
  const bf16* vgl = qkv + ((((size_t)b * 3 + 2) * Hh + h) * Nn) * DHd;

  // staging coords: 32x96 tile = 384 16B chunks, 3 per thread
  int srow[3], scol[3];
#pragma unroll
  for (int i = 0; i < 3; ++i) {
    int cid = threadIdx.x + i * 128;
    srow[i] = cid / 12;
    scol[i] = cid % 12;
  }

  // Q fragments: lane = row (query), 3 chunks of K=32 along DH
  const int qrow = q0 + wave * 16 + (lane & 15);
  v16bf qf[3];
#pragma unroll
  for (int c = 0; c < 3; ++c)
    qf[c] = load_frag(qgl + (size_t)qrow * DHd + c * 32, kb);

  v8f acc[6];
#pragma unroll
  for (int j = 0; j < 6; ++j) acc[j] = v8f_zero();
  float mrow[8], lrow[8];
#pragma unroll
  for (int r = 0; r < 8; ++r) { mrow[r] = -3.0e38f; lrow[r] = 0.0f; }

  for (int kt = 0; kt < Nn / 32; ++kt) {
    const bf16* kg = kgl + (size_t)kt * 32 * DHd;
    const bf16* vg = vgl + (size_t)kt * 32 * DHd;
    // issue all 6 staging loads up front (distinct temps -> overlapped)
    v8bf kc0 = *(const v8bf*)(kg + (size_t)srow[0] * DHd + scol[0] * 8);
    v8bf kc1 = *(const v8bf*)(kg + (size_t)srow[1] * DHd + scol[1] * 8);
    v8bf kc2 = *(const v8bf*)(kg + (size_t)srow[2] * DHd + scol[2] * 8);
    v8bf vc0 = *(const v8bf*)(vg + (size_t)srow[0] * DHd + scol[0] * 8);
    v8bf vc1 = *(const v8bf*)(vg + (size_t)srow[1] * DHd + scol[1] * 8);
    v8bf vc2 = *(const v8bf*)(vg + (size_t)srow[2] * DHd + scol[2] * 8);
    // K tile: straight 16B stores
    *(v8bf*)&Ks[srow[0]][scol[0] * 8] = kc0;
    *(v8bf*)&Ks[srow[1]][scol[1] * 8] = kc1;
    *(v8bf*)&Ks[srow[2]][scol[2] * 8] = kc2;
    // V^T tile: lane-local scatter of each 8-elem chunk
#pragma unroll
    for (int e = 0; e < 8; ++e) {
      Vt[scol[0] * 8 + e][srow[0]] = vc0[e];
      Vt[scol[1] * 8 + e][srow[1]] = vc1[e];
      Vt[scol[2] * 8 + e][srow[2]] = vc2[e];
    }
    __syncthreads();

    // ---- S = Q K^T : two 16-key halves, 3 k-chunks each ----
    v8f s0 = v8f_zero(), s1 = v8f_zero();
#pragma unroll
    for (int c = 0; c < 3; ++c) {
      v16bf kf = load_frag(&Ks[0 + (lane & 15)][c * 32], kb);
      s0 = wmma_bf16(qf[c], kf, s0);
    }
#pragma unroll
    for (int c = 0; c < 3; ++c) {
      v16bf kf = load_frag(&Ks[16 + (lane & 15)][c * 32], kb);
      s1 = wmma_bf16(qf[c], kf, s1);
    }

    // ---- online softmax: rows live in vgpr r, cols across 16-lane half ----
#pragma unroll
    for (int r = 0; r < 8; ++r) {
      float a0 = s0[r] * kScale;
      float a1 = s1[r] * kScale;
      float tm = fmaxf(a0, a1);
#pragma unroll
      for (int d = 1; d < 16; d <<= 1) tm = fmaxf(tm, __shfl_xor(tm, d, 32));
      float mnew  = fmaxf(mrow[r], tm);
      float alpha = __expf(mrow[r] - mnew);
      mrow[r] = mnew;
      float p0 = __expf(a0 - mnew);
      float p1 = __expf(a1 - mnew);
      float rs = p0 + p1;
#pragma unroll
      for (int d = 1; d < 16; d <<= 1) rs += __shfl_xor(rs, d, 32);
      lrow[r] = lrow[r] * alpha + rs;
#pragma unroll
      for (int j = 0; j < 6; ++j) acc[j][r] *= alpha;
      int m = r + ((lane >> 4) << 3);
      Ps[wave][m][lane & 15]        = (bf16)p0;
      Ps[wave][m][16 + (lane & 15)] = (bf16)p1;
    }
    asm volatile("s_wait_dscnt 0" ::: "memory");  // same-wave P write -> read

    // ---- O += P V : A = P (16x32), B cols = dh via V^T ----
    v16bf pf = load_frag(&Ps[wave][lane & 15][0], kb);
#pragma unroll
    for (int j = 0; j < 6; ++j) {
      v16bf vf = load_frag(&Vt[j * 16 + (lane & 15)][0], kb);
      acc[j] = wmma_bf16(pf, vf, acc[j]);
    }
    __syncthreads();  // before next tile overwrites Ks/Vt
  }

  // ---- normalize + store bf16 to ao[b][n][h*DH + dh] ----
#pragma unroll
  for (int j = 0; j < 6; ++j)
#pragma unroll
    for (int r = 0; r < 8; ++r) {
      int m = r + ((lane >> 4) << 3);
      int qn = q0 + wave * 16 + m;
      float o = acc[j][r] / lrow[r];
      ao[((size_t)b * Nn + qn) * NHd + h * DHd + j * 16 + (lane & 15)] = (bf16)o;
    }
}

// ------------------------------------------------------------------
extern "C" void kernel_launch(void* const* d_in, const int* in_sizes, int n_in,
                              void* d_out, int out_size, void* d_ws,
                              size_t ws_size, hipStream_t stream) {
  (void)in_sizes; (void)n_in; (void)out_size; (void)ws_size;
  const float* x    = (const float*)d_in[0];
  const float* f1   = (const float*)d_in[1];
  const float* f2   = (const float*)d_in[2];
  const float* f3   = (const float*)d_in[3];
  const float* Wqkv = (const float*)d_in[4];
  const float* Wout = (const float*)d_in[5];
  const float* bout = (const float*)d_in[6];
  float* out = (float*)d_out;

  char* ws = (char*)d_ws;
  bf16* xb    = (bf16*)(ws + OFF_XB);
  bf16* wqkvt = (bf16*)(ws + OFF_WQKVT);
  bf16* woutt = (bf16*)(ws + OFF_WOUTT);
  bf16* qkv   = (bf16*)(ws + OFF_QKV);
  bf16* ao    = (bf16*)(ws + OFF_AO);

  // 1) precision conversion / weight transposition
  cvt_f32_to_bf16<<<(Mrows * DIMd + 255) / 256, 256, 0, stream>>>(
      x, xb, Mrows * DIMd);
  transpose_cvt<<<(DIMd * QKVC + 255) / 256, 256, 0, stream>>>(
      Wqkv, wqkvt, DIMd, QKVC);
  transpose_cvt<<<(NHd * DIMd + 255) / 256, 256, 0, stream>>>(
      Wout, woutt, NHd, DIMd);

  // 2) QKV projection -> head-major bf16   (grid: 64 * 36 tiles)
  gemm_bf16_kernel<<<(Mrows / 128) * (QKVC / 128), 256, 0, stream>>>(
      xb, wqkvt, Mrows, DIMd, QKVC, /*mode=*/0, qkv, nullptr, nullptr);

  // 3) RoPE on q,k
  rope_kernel<<<(Bb * Hh * Nn * 3 * 16 + 255) / 256, 256, 0, stream>>>(
      qkv, f1, f2, f3);

  // 4) flash attention
  attn_kernel<<<Bb * Hh * (Nn / 64), 128, 0, stream>>>(qkv, ao);

  // 5) output projection + bias (f32 out)   (grid: 64 * 12 tiles)
  gemm_bf16_kernel<<<(Mrows / 128) * (DIMd / 128), 256, 0, stream>>>(
      ao, woutt, Mrows, DIMd, DIMd, /*mode=*/1, nullptr, out, bout);
}
